// WindowAttention2D_19885698580536
// MI455X (gfx1250) — matrix operand order repaired and checked by reading, not compiled, not run on hardware
//
#include <hip/hip_runtime.h>
#include <hip/hip_bf16.h>
#include <cstddef>
#include <cstdint>

// ---------------------------------------------------------------------------
// WindowAttention2D on MI455X (gfx1250, wave32, WMMA 16x16x32 f16)
//
//   K1: qkv projection  W(512x512) @ X(512x4096) per image, window-permuted
//       f16 store into workspace as [proj][b][e][win][t]
//   K2: per-(window,head) attention: S = Q K^T, softmax, O = P V
//       (V staged via GLOBAL_LOAD_ASYNC_TO_LDS_B128, ASYNCcnt-tracked)
//   K3: out projection + bias, inverse window permutation, f32 output
// ---------------------------------------------------------------------------

typedef __attribute__((ext_vector_type(16))) _Float16 v16h;
typedef __attribute__((ext_vector_type(8)))  _Float16 v8h;
typedef __attribute__((ext_vector_type(8)))  float    v8f;
typedef __attribute__((ext_vector_type(4)))  float    v4f;

#define E_DIM 512
#define HWPIX 4096          // 64*64 pixels per image (halves-stride per e row)
#define BM 128
#define BN 128
#define BK 32
#define LSA 40              // LDS row stride (halves) for BK=32 tiles; 80B
#define AST 72              // LDS row stride (halves) for 64-wide tiles; 144B

__device__ __forceinline__ v8f wmma16(v16h a, v16h b, v8f c) {
  return __builtin_amdgcn_wmma_f32_16x16x32_f16(
      false, a, false, b, (short)0, c, false, false);
}

// A fragment (16x32 f16): LDS row-major [m][k]. lane = h*16+r holds row r,
// halves 0..7 = K(h*8..+7), halves 8..15 = K(16+h*8..+7). Two 16B reads.
__device__ __forceinline__ v16h frag_a_row(const _Float16* base, int stride) {
  const int lane = threadIdx.x & 31;
  const int r = lane & 15, hh = lane >> 4;
  const _Float16* p = base + r * stride;
  union { v16h v; v8h h[2]; } u;
  u.h[0] = *(const v8h*)(p + hh * 8);
  u.h[1] = *(const v8h*)(p + 16 + hh * 8);
  return u.v;
}

// B fragment (32x16 f16): LDS holds B^T [n][k]. lane = h*16+n holds column n,
// halves 0..7 = K(h*8..+7), halves 8..15 = K(16+h*8..+7). Two 16B reads.
__device__ __forceinline__ v16h frag_b_rowT(const _Float16* base, int stride) {
  const int lane = threadIdx.x & 31;
  const int n = lane & 15, hh = lane >> 4;
  const _Float16* p = base + n * stride;
  union { v16h v; v8h h[2]; } u;
  u.h[0] = *(const v8h*)(p + hh * 8);
  u.h[1] = *(const v8h*)(p + 16 + hh * 8);
  return u.v;
}

// pack two f32 -> one dword of two f16 (compiler emits v_cvt_pk_f16_f32)
__device__ __forceinline__ uint32_t pk2h(float a, float b) {
  union { _Float16 h[2]; uint32_t u; } p;
  p.h[0] = (_Float16)a; p.h[1] = (_Float16)b;
  return p.u;
}

// ---------------------------------------------------------------------------
// Kernel 1: QKV projection.  grid=(32,4,24), block=256 (8 waves, 2x4).
// ---------------------------------------------------------------------------
__global__ __launch_bounds__(256)
void wa2d_qkv_proj(const float* __restrict__ q, const float* __restrict__ k,
                   const float* __restrict__ v, const float* __restrict__ w,
                   const float* __restrict__ bias, _Float16* __restrict__ out) {
  __shared__ _Float16 ldsA[BM * LSA];
  __shared__ _Float16 ldsB[BN * LSA];

  const int n0 = blockIdx.x * BN;
  const int m0 = blockIdx.y * BM;
  const int z = blockIdx.z;
  const int proj = z >> 3, b = z & 7;

  const float* x  = (proj == 0) ? q : (proj == 1) ? k : v;
  const float* Wp = w + (size_t)proj * E_DIM * E_DIM;
  const float* bp = bias + proj * E_DIM;
  _Float16* outp  = out + (size_t)proj * 8 * E_DIM * HWPIX;

  const int tid = threadIdx.x;
  const int wave = tid >> 5, lane = tid & 31;
  const int wm = wave >> 2, wn = wave & 3;      // wave covers 64(m) x 32(n)
  const int ln = lane & 15, lh = lane >> 4;

  v8f acc[4][2] = {};

  for (int k0 = 0; k0 < E_DIM; k0 += BK) {
    // ---- stage A: weight tile [128 m][32 k], f32->f16, two b128 LDS stores
    {
      const int row = tid >> 1;
      const int seg = (tid & 1) * 16;
      const float* g = Wp + (size_t)(m0 + row) * E_DIM + k0 + seg;
      v4f f0 = *(const v4f*)(g + 0);
      v4f f1 = *(const v4f*)(g + 4);
      v4f f2 = *(const v4f*)(g + 8);
      v4f f3 = *(const v4f*)(g + 12);
      union { v8h v; _Float16 h[8]; } h0, h1;
#pragma unroll
      for (int j = 0; j < 4; ++j) {
        h0.h[j] = (_Float16)f0[j]; h0.h[4 + j] = (_Float16)f1[j];
        h1.h[j] = (_Float16)f2[j]; h1.h[4 + j] = (_Float16)f3[j];
      }
      _Float16* l = ldsA + row * LSA + seg;
      *(v8h*)(l)     = h0.v;
      *(v8h*)(l + 8) = h1.v;
    }
    // ---- stage B: activation [32 k][128 n]; each thread stages k-pair so the
    //      [n][k] transpose packs into b32 LDS stores
    {
      const int kk = (tid >> 4) * 2;           // 0,2,...,30
      const int nn = (tid & 15) * 8;           // 0..120
      const float* g0 = x + ((size_t)b * E_DIM + k0 + kk) * HWPIX + n0 + nn;
      const float* g1 = g0 + HWPIX;
      if (k0 + BK < E_DIM) __builtin_prefetch(g0 + (size_t)BK * HWPIX, 0, 1);
      v4f a0 = *(const v4f*)(g0);
      v4f a1 = *(const v4f*)(g0 + 4);
      v4f c0 = *(const v4f*)(g1);
      v4f c1 = *(const v4f*)(g1 + 4);
#pragma unroll
      for (int j = 0; j < 4; ++j) {
        *(uint32_t*)&ldsB[(nn + j) * LSA + kk]     = pk2h(a0[j], c0[j]);
        *(uint32_t*)&ldsB[(nn + 4 + j) * LSA + kk] = pk2h(a1[j], c1[j]);
      }
    }
    __syncthreads();

    v16h af[4], bf[2];
#pragma unroll
    for (int mi = 0; mi < 4; ++mi)
      af[mi] = frag_a_row(ldsA + (wm * 64 + mi * 16) * LSA, LSA);
#pragma unroll
    for (int ni = 0; ni < 2; ++ni)
      bf[ni] = frag_b_rowT(ldsB + (wn * 32 + ni * 16) * LSA, LSA);
#pragma unroll
    for (int mi = 0; mi < 4; ++mi)
#pragma unroll
      for (int ni = 0; ni < 2; ++ni)
        acc[mi][ni] = wmma16(af[mi], bf[ni], acc[mi][ni]);
    __syncthreads();
  }

  // ---- epilogue: bias + window-permuted f16 store
#pragma unroll
  for (int mi = 0; mi < 4; ++mi) {
#pragma unroll
    for (int ni = 0; ni < 2; ++ni) {
      const int nglob = n0 + wn * 32 + ni * 16 + ln;
      const int hi = nglob >> 6, wi = nglob & 63;
      const int win = ((hi >> 3) << 3) | (wi >> 3);
      const int tt  = ((hi & 7) << 3) | (wi & 7);
#pragma unroll
      for (int r = 0; r < 8; ++r) {
        const int e = m0 + wm * 64 + mi * 16 + lh * 8 + r;
        const float val = acc[mi][ni][r] + bp[e];
        outp[(((size_t)b * E_DIM + e) * 64 + win) * 64 + tt] = (_Float16)val;
      }
    }
  }
}

// ---------------------------------------------------------------------------
// Kernel 2: per-(window, head) attention. grid=(8,512), block=128 (4 waves).
// Q,K staged transposed [t][d] with packed b32 stores; V staged [d][t] via
// GLOBAL_LOAD_ASYNC_TO_LDS_B128 (ASYNCcnt). O returns through LDS for
// coalesced 16B global stores.
// ---------------------------------------------------------------------------
__global__ __launch_bounds__(128)
void wa2d_attn(const _Float16* __restrict__ qkv, _Float16* __restrict__ o) {
  __shared__ _Float16 sQ[64 * AST];
  __shared__ _Float16 sK[64 * AST];
  __shared__ _Float16 sV[64 * AST];
  __shared__ _Float16 sP[64 * AST];

  const int head = blockIdx.x;
  const int nw = blockIdx.y;
  const int b = nw >> 6, wl = nw & 63;
  const int tid = threadIdx.x;
  const int wave = tid >> 5, lane = tid & 31;
  const int ln = lane & 15, lh = lane >> 4;

  const size_t plane = (size_t)8 * E_DIM * HWPIX;
  const size_t base = (((size_t)b * E_DIM + head * 64) * 64 + wl) * 64;
  const _Float16* gq = qkv + base;
  const _Float16* gk = qkv + plane + base;
  const _Float16* gv = qkv + 2 * plane + base;

  // ---- V: async DMA copy straight into LDS in natural [d][t] layout
  {
    const int dv = tid >> 1;
    const int segv = (tid & 1) * 32;
    const _Float16* vg = gv + (size_t)dv * HWPIX + segv;
    const uint32_t vdst = (uint32_t)(uintptr_t)&sV[dv * AST + segv];
#pragma unroll
    for (int c = 0; c < 4; ++c)
      asm volatile("global_load_async_to_lds_b128 %0, %1, off"
                   :: "v"(vdst + c * 16),
                      "v"((unsigned long long)(uintptr_t)(vg + c * 8))
                   : "memory");
  }
  // ---- Q,K: stage transposed to [t][d]; even-d pairs pack into b32 stores
  {
    const int d0 = (tid >> 2) * 2;           // 0,2,...,62
    const int seg = (tid & 3) * 16;          // 0,16,32,48
    const _Float16* q0 = gq + (size_t)d0 * HWPIX + seg;
    const _Float16* q1 = q0 + HWPIX;
    const _Float16* ka = gk + (size_t)d0 * HWPIX + seg;
    const _Float16* kb = ka + HWPIX;
    v8h qa0 = *(const v8h*)(q0), qa1 = *(const v8h*)(q0 + 8);
    v8h qb0 = *(const v8h*)(q1), qb1 = *(const v8h*)(q1 + 8);
    v8h ka0 = *(const v8h*)(ka), ka1 = *(const v8h*)(ka + 8);
    v8h kb0 = *(const v8h*)(kb), kb1 = *(const v8h*)(kb + 8);
#pragma unroll
    for (int j = 0; j < 8; ++j) {
      union { _Float16 h[2]; uint32_t u; } p;
      p.h[0] = qa0[j]; p.h[1] = qb0[j];
      *(uint32_t*)&sQ[(seg + j) * AST + d0] = p.u;
      p.h[0] = qa1[j]; p.h[1] = qb1[j];
      *(uint32_t*)&sQ[(seg + 8 + j) * AST + d0] = p.u;
      p.h[0] = ka0[j]; p.h[1] = kb0[j];
      *(uint32_t*)&sK[(seg + j) * AST + d0] = p.u;
      p.h[0] = ka1[j]; p.h[1] = kb1[j];
      *(uint32_t*)&sK[(seg + 8 + j) * AST + d0] = p.u;
    }
  }
  asm volatile("s_wait_asynccnt 0x0" ::: "memory");
  __syncthreads();

  const int mrow = wave * 16;

  // ---- S = Q K^T  (wave: 16 q-rows x 64 k-cols)
  v8f s[4] = {};
#pragma unroll
  for (int k0 = 0; k0 < 64; k0 += 32) {
    v16h a = frag_a_row(sQ + mrow * AST + k0, AST);
#pragma unroll
    for (int ni = 0; ni < 4; ++ni) {
      v16h bb = frag_b_rowT(sK + (ni * 16) * AST + k0, AST);
      s[ni] = wmma16(a, bb, s[ni]);
    }
  }

  // ---- softmax across 64-wide rows (16-lane butterflies stay in lane group)
  const float scale = 0.125f;   // 1/sqrt(64)
  float rsum[8];
#pragma unroll
  for (int r = 0; r < 8; ++r) {
    float m = -3.0e38f;
#pragma unroll
    for (int ni = 0; ni < 4; ++ni) m = fmaxf(m, s[ni][r]);
#pragma unroll
    for (int off = 1; off < 16; off <<= 1) m = fmaxf(m, __shfl_xor(m, off, 32));
    float sum = 0.0f;
#pragma unroll
    for (int ni = 0; ni < 4; ++ni) {
      const float p = __expf((s[ni][r] - m) * scale);
      s[ni][r] = p;
      sum += p;
    }
#pragma unroll
    for (int off = 1; off < 16; off <<= 1) sum += __shfl_xor(sum, off, 32);
    rsum[r] = sum;
  }

  // ---- P -> LDS (C-layout regs to A-fragment layout), wave-private rows
#pragma unroll
  for (int ni = 0; ni < 4; ++ni)
#pragma unroll
    for (int r = 0; r < 8; ++r)
      sP[(mrow + lh * 8 + r) * AST + ni * 16 + ln] = (_Float16)(s[ni][r] / rsum[r]);
  __syncthreads();

  // ---- O = P V  (wave: 16 q-rows x 64 head-dims)
  v8f oacc[4] = {};
#pragma unroll
  for (int k0 = 0; k0 < 64; k0 += 32) {
    v16h a = frag_a_row(sP + mrow * AST + k0, AST);
#pragma unroll
    for (int ni = 0; ni < 4; ++ni) {
      v16h bb = frag_b_rowT(sV + (ni * 16) * AST + k0, AST);
      oacc[ni] = wmma16(a, bb, oacc[ni]);
    }
  }

  // ---- O: pack rows (contiguous in t) into b128 LDS stores (reuse sQ, dead
  //      since the softmax barrier), then coalesced 16B global writeback
#pragma unroll
  for (int ni = 0; ni < 4; ++ni) {
    const int d = ni * 16 + ln;
    union { v8h v; _Float16 h[8]; } pk;
#pragma unroll
    for (int r = 0; r < 8; ++r) pk.h[r] = (_Float16)oacc[ni][r];
    *(v8h*)(sQ + d * AST + mrow + lh * 8) = pk.v;
  }
  __syncthreads();
  {
    const int d = tid >> 1;
    const int seg = (tid & 1) * 32;
    _Float16* og = o + base + (size_t)d * HWPIX + seg;
#pragma unroll
    for (int c = 0; c < 4; ++c)
      *(v8h*)(og + c * 8) = *(const v8h*)(sQ + d * AST + seg + c * 8);
  }
}

// ---------------------------------------------------------------------------
// Kernel 3: output projection + bias, inverse permutation, f32 out.
// grid=(32,4,8), block=256.
// ---------------------------------------------------------------------------
__global__ __launch_bounds__(256)
void wa2d_out_proj(const _Float16* __restrict__ xin, const float* __restrict__ w,
                   const float* __restrict__ bias, float* __restrict__ out) {
  __shared__ _Float16 ldsA[BM * LSA];
  __shared__ _Float16 ldsB[BN * LSA];

  const int n0 = blockIdx.x * BN;
  const int m0 = blockIdx.y * BM;
  const int b = blockIdx.z;

  const int tid = threadIdx.x;
  const int wave = tid >> 5, lane = tid & 31;
  const int wm = wave >> 2, wn = wave & 3;
  const int ln = lane & 15, lh = lane >> 4;

  v8f acc[4][2] = {};

  for (int k0 = 0; k0 < E_DIM; k0 += BK) {
    // ---- stage A: out_proj_w tile (f32->f16), b128 LDS stores
    {
      const int row = tid >> 1;
      const int seg = (tid & 1) * 16;
      const float* g = w + (size_t)(m0 + row) * E_DIM + k0 + seg;
      v4f f0 = *(const v4f*)(g + 0);
      v4f f1 = *(const v4f*)(g + 4);
      v4f f2 = *(const v4f*)(g + 8);
      v4f f3 = *(const v4f*)(g + 12);
      union { v8h v; _Float16 h[8]; } h0, h1;
#pragma unroll
      for (int j = 0; j < 4; ++j) {
        h0.h[j] = (_Float16)f0[j]; h0.h[4 + j] = (_Float16)f1[j];
        h1.h[j] = (_Float16)f2[j]; h1.h[4 + j] = (_Float16)f3[j];
      }
      _Float16* l = ldsA + row * LSA + seg;
      *(v8h*)(l)     = h0.v;
      *(v8h*)(l + 8) = h1.v;
    }
    // ---- stage B: attention-out tile (f16); k-pair staging -> b32 stores
    {
      const int kk = (tid >> 4) * 2;
      const int nn = (tid & 15) * 8;
      const _Float16* g0 = xin + ((size_t)b * E_DIM + k0 + kk) * HWPIX + n0 + nn;
      const _Float16* g1 = g0 + HWPIX;
      if (k0 + BK < E_DIM) __builtin_prefetch(g0 + (size_t)BK * HWPIX, 0, 1);
      v8h r0 = *(const v8h*)(g0);
      v8h r1 = *(const v8h*)(g1);
#pragma unroll
      for (int j = 0; j < 8; ++j) {
        union { _Float16 h[2]; uint32_t u; } p;
        p.h[0] = r0[j]; p.h[1] = r1[j];
        *(uint32_t*)&ldsB[(nn + j) * LSA + kk] = p.u;
      }
    }
    __syncthreads();

    v16h af[4], bf[2];
#pragma unroll
    for (int mi = 0; mi < 4; ++mi)
      af[mi] = frag_a_row(ldsA + (wm * 64 + mi * 16) * LSA, LSA);
#pragma unroll
    for (int ni = 0; ni < 2; ++ni)
      bf[ni] = frag_b_rowT(ldsB + (wn * 32 + ni * 16) * LSA, LSA);
#pragma unroll
    for (int mi = 0; mi < 4; ++mi)
#pragma unroll
      for (int ni = 0; ni < 2; ++ni)
        acc[mi][ni] = wmma16(af[mi], bf[ni], acc[mi][ni]);
    __syncthreads();
  }

  // ---- epilogue: bias + inverse window permutation, f32 store to (B,E,H,W)
#pragma unroll
  for (int mi = 0; mi < 4; ++mi) {
#pragma unroll
    for (int ni = 0; ni < 2; ++ni) {
      const int nglob = n0 + wn * 32 + ni * 16 + ln;    // win*64 + t
      const int win = nglob >> 6, tt = nglob & 63;
      const int hi = ((win >> 3) << 3) | (tt >> 3);
      const int wi = ((win & 7) << 3) | (tt & 7);
#pragma unroll
      for (int r = 0; r < 8; ++r) {
        const int e = m0 + wm * 64 + mi * 16 + lh * 8 + r;
        out[(((size_t)b * E_DIM + e) * 64 + hi) * 64 + wi] = acc[mi][ni][r] + bias[e];
      }
    }
  }
}

// ---------------------------------------------------------------------------
extern "C" void kernel_launch(void* const* d_in, const int* in_sizes, int n_in,
                              void* d_out, int out_size, void* d_ws, size_t ws_size,
                              hipStream_t stream) {
  const float* q     = (const float*)d_in[0];
  const float* k     = (const float*)d_in[1];
  const float* v     = (const float*)d_in[2];
  const float* in_w  = (const float*)d_in[3];
  const float* in_b  = (const float*)d_in[4];
  const float* out_w = (const float*)d_in[5];
  const float* out_b = (const float*)d_in[6];
  float* out = (float*)d_out;

  // workspace: qkv f16 [3][8][512][4096] (96 MiB) + attn-out f16 (32 MiB)
  _Float16* qkv = (_Float16*)d_ws;
  _Float16* ows = qkv + (size_t)3 * 8 * E_DIM * HWPIX;

  dim3 gA(HWPIX / BN, E_DIM / BM, 24);   // 32 x 4 x (3 proj * 8 images)
  wa2d_qkv_proj<<<gA, 256, 0, stream>>>(q, k, v, in_w, in_b, qkv);

  dim3 gB(8, 512, 1);                    // heads x windows
  wa2d_attn<<<gB, 128, 0, stream>>>(qkv, ows);

  dim3 gC(HWPIX / BN, E_DIM / BM, 8);    // 32 x 4 x images
  wa2d_out_proj<<<gC, 256, 0, stream>>>(ows, out_w, out_b, out);
}
